// SourceAnchoredBlock_33681133535740
// MI455X (gfx1250) — matrix operand order repaired and checked by reading, not compiled, not run on hardware
//
#include <hip/hip_runtime.h>
#include <hip/hip_bf16.h>
#include <cstdint>
#include <cstddef>

// ---------------- CDNA5 WMMA types ----------------
typedef __attribute__((ext_vector_type(16))) __bf16   v16bf;
typedef __attribute__((ext_vector_type(8)))  float    v8f;
typedef __attribute__((ext_vector_type(4)))  unsigned uint4v;

union FragBF { v16bf v; uint4v q[2]; };

__device__ __forceinline__ __bf16 f2bf(float f) {
  union { float f; unsigned u; } x; x.f = f;
  unsigned r = x.u + 0x7FFFu + ((x.u >> 16) & 1u);   // round-to-nearest-even
  unsigned short h = (unsigned short)(r >> 16);
  return __builtin_bit_cast(__bf16, h);
}

__device__ __forceinline__ v8f v8f_zero() {
  v8f z;
#pragma unroll
  for (int e = 0; e < 8; ++e) z[e] = 0.f;
  return z;
}

__device__ __forceinline__ v8f wmma_bf16(v16bf a, v16bf b, v8f c) {
  // D = A(16x32 bf16) * B(32x16 bf16) + C(16x16 f32)
  return __builtin_amdgcn_wmma_f32_16x16x32_bf16(false, a, false, b, (short)0, c,
                                                 false, false);
}

// ---------------- step-embedding add ----------------
__global__ __launch_bounds__(256)
void sab_add_step(const float* __restrict__ hidden, const float* __restrict__ emb,
                  const int* __restrict__ step, float* __restrict__ out,
                  int total, int D) {
  int i = blockIdx.x * 256 + threadIdx.x;
  if (i >= total) return;
  int s = step[0];
  s = s < 0 ? 0 : (s > 19 ? 19 : s);
  out[i] = hidden[i] + emb[s * D + (i % D)];
}

// ---------------- generic bf16-WMMA GEMM: C = A @ W + bias (opt GELU) ----------------
#define GT_M 128
#define GT_N 128
#define GT_K 32
#define GP   40   // padded LDS row length (bf16 elems); 80B rows, 16B aligned

__global__ __launch_bounds__(256)
void sab_gemm_wmma(const float* __restrict__ A, const float* __restrict__ W,
                   const float* __restrict__ bias, float* __restrict__ C,
                   int M, int N, int K, int gelu) {
  __shared__ __align__(16) __bf16 sA[GT_M * GP];
  __shared__ __align__(16) __bf16 sB[GT_N * GP];

  const int tid  = threadIdx.x;
  const int lane = tid & 31;
  const int wave = tid >> 5;
  const int m0b  = blockIdx.y * GT_M;
  const int n0b  = blockIdx.x * GT_N;
  const int wm   = wave & 3;     // 32-row group
  const int wn   = wave >> 2;    // 64-col group
  const int r    = lane & 15;
  const int g    = lane >> 4;    // half-wave select

  v8f acc[2][4];
#pragma unroll
  for (int i = 0; i < 2; ++i)
#pragma unroll
    for (int j = 0; j < 4; ++j) acc[i][j] = v8f_zero();

  for (int k0 = 0; k0 < K; k0 += GT_K) {
    __syncthreads();
    {   // A tile: 128 x 32, f32 -> bf16
      const int tr = tid >> 1;
      const int tc = (tid & 1) * 16;
      const float* src = A + (size_t)(m0b + tr) * K + (k0 + tc);
      __bf16* dst = &sA[tr * GP + tc];
#pragma unroll
      for (int j = 0; j < 16; j += 4) {
        float4 f = *(const float4*)(src + j);
        dst[j]     = f2bf(f.x); dst[j + 1] = f2bf(f.y);
        dst[j + 2] = f2bf(f.z); dst[j + 3] = f2bf(f.w);
      }
    }
    {   // B tile transposed: sB[n][k] = W[k0+k][n0b+n]
      const int kr = tid >> 3;
      const int nc = (tid & 7) * 16;
      const float* src = W + (size_t)(k0 + kr) * N + (n0b + nc);
#pragma unroll
      for (int j = 0; j < 16; j += 4) {
        float4 f = *(const float4*)(src + j);
        sB[(nc + j    ) * GP + kr] = f2bf(f.x);
        sB[(nc + j + 1) * GP + kr] = f2bf(f.y);
        sB[(nc + j + 2) * GP + kr] = f2bf(f.z);
        sB[(nc + j + 3) * GP + kr] = f2bf(f.w);
      }
    }
    __syncthreads();

    FragBF af[2], bfr[4];
#pragma unroll
    for (int i = 0; i < 2; ++i) {   // A frag: lane row = r, K pairs at koff=8g
      const unsigned* row = (const unsigned*)&sA[(wm * 32 + i * 16 + r) * GP];
      af[i].q[0] = *(const uint4v*)(row + g * 4);
      af[i].q[1] = *(const uint4v*)(row + 8 + g * 4);
    }
#pragma unroll
    for (int j = 0; j < 4; ++j) {   // B frag: lane col = r, K pairs at koff=8g
      const unsigned* row = (const unsigned*)&sB[(wn * 64 + j * 16 + r) * GP];
      bfr[j].q[0] = *(const uint4v*)(row + g * 4);
      bfr[j].q[1] = *(const uint4v*)(row + 8 + g * 4);
    }
#pragma unroll
    for (int i = 0; i < 2; ++i)
#pragma unroll
      for (int j = 0; j < 4; ++j)
        acc[i][j] = wmma_bf16(af[i].v, bfr[j].v, acc[i][j]);
  }

#pragma unroll
  for (int i = 0; i < 2; ++i) {
    const int mrow = m0b + wm * 32 + i * 16 + g * 8;
#pragma unroll
    for (int j = 0; j < 4; ++j) {
      const int ncol = n0b + wn * 64 + j * 16 + r;
      const float bv = bias ? bias[ncol] : 0.f;
#pragma unroll
      for (int e = 0; e < 8; ++e) {
        float v = acc[i][j][e] + bv;
        if (gelu) v = 0.5f * v * (1.f + erff(v * 0.70710678118654752f));
        C[(size_t)(mrow + e) * N + ncol] = v;
      }
    }
  }
}

// ---------------- flash attention (WMMA scores + PV), opt rel-pos bias ----------------
__global__ __launch_bounds__(128)
void sab_attn_wmma(const float* __restrict__ qb, int qstride,
                   const float* __restrict__ kb, int kstride,
                   const float* __restrict__ vb, int vstride,
                   const float* __restrict__ relbias,
                   float* __restrict__ out, int ostride, int S) {
  __shared__ __align__(16) __bf16 sQ[64][64];        // [q][d]
  __shared__ __align__(16) __bf16 sK[32][64];        // [k][d]
  __shared__ __align__(16) __bf16 sV[64][32];        // [d][k] (transposed)
  __shared__ __align__(16) __bf16 sP[4][16][32];     // per-wave probs

  const int tid  = threadIdx.x;
  const int lane = tid & 31;
  const int wave = tid >> 5;
  const int bI = blockIdx.z, hI = blockIdx.y;
  const int q0 = blockIdx.x * 64;
  const int r = lane & 15, g = lane >> 4;

  const float* qp = qb + (size_t)bI * S * qstride + hI * 64;
  const float* kp = kb + (size_t)bI * S * kstride + hI * 64;
  const float* vp = vb + (size_t)bI * S * vstride + hI * 64;

  // stage Q (64 rows x 64 dims)
  for (int i = tid; i < 64 * 16; i += 128) {
    int qq = i >> 4, dd = (i & 15) * 4;
    float4 f = *(const float4*)(qp + (size_t)(q0 + qq) * qstride + dd);
    sQ[qq][dd]     = f2bf(f.x); sQ[qq][dd + 1] = f2bf(f.y);
    sQ[qq][dd + 2] = f2bf(f.z); sQ[qq][dd + 3] = f2bf(f.w);
  }
  __syncthreads();

  FragBF qa[2];
  {
    const unsigned* row = (const unsigned*)&sQ[wave * 16 + r][0];
    qa[0].q[0] = *(const uint4v*)(row + g * 4);
    qa[0].q[1] = *(const uint4v*)(row + 8 + g * 4);
    qa[1].q[0] = *(const uint4v*)(row + 16 + g * 4);
    qa[1].q[1] = *(const uint4v*)(row + 24 + g * 4);
  }

  float mrun[8], lrun[8];
  v8f o[4];
#pragma unroll
  for (int j = 0; j < 8; ++j) { mrun[j] = -3.0e38f; lrun[j] = 0.f; }
#pragma unroll
  for (int nt = 0; nt < 4; ++nt) o[nt] = v8f_zero();

  for (int kv0 = 0; kv0 < S; kv0 += 32) {
    __syncthreads();
    {   // K tile 32x64
      int kk = tid >> 2, dd = (tid & 3) * 16;
      const float* src = kp + (size_t)(kv0 + kk) * kstride + dd;
#pragma unroll
      for (int j = 0; j < 16; j += 4) {
        float4 f = *(const float4*)(src + j);
        sK[kk][dd + j]     = f2bf(f.x); sK[kk][dd + j + 1] = f2bf(f.y);
        sK[kk][dd + j + 2] = f2bf(f.z); sK[kk][dd + j + 3] = f2bf(f.w);
      }
    }
    {   // V tile 32x64 -> sV[d][k]
      int kk = tid >> 2, dd = (tid & 3) * 16;
      const float* src = vp + (size_t)(kv0 + kk) * vstride + dd;
#pragma unroll
      for (int j = 0; j < 16; j += 4) {
        float4 f = *(const float4*)(src + j);
        sV[dd + j][kk]     = f2bf(f.x); sV[dd + j + 1][kk] = f2bf(f.y);
        sV[dd + j + 2][kk] = f2bf(f.z); sV[dd + j + 3][kk] = f2bf(f.w);
      }
    }
    __syncthreads();

    // scores: Q(16x64) x K^T(64x32) -> two 16x16 f32 tiles
    v8f sc[2];
    sc[0] = v8f_zero(); sc[1] = v8f_zero();
#pragma unroll
    for (int nt = 0; nt < 2; ++nt) {
      const unsigned* row = (const unsigned*)&sK[nt * 16 + r][0];
#pragma unroll
      for (int c = 0; c < 2; ++c) {
        FragBF kf;
        kf.q[0] = *(const uint4v*)(row + c * 16 + g * 4);
        kf.q[1] = *(const uint4v*)(row + c * 16 + 8 + g * 4);
        sc[nt] = wmma_bf16(qa[c].v, kf.v, sc[nt]);
      }
    }

    // online softmax (rows j+8g, cols across 16-lane half-groups)
#pragma unroll
    for (int j = 0; j < 8; ++j) {
      float s0 = sc[0][j] * 0.125f;   // HD^-0.5 = 1/8
      float s1 = sc[1][j] * 0.125f;
      if (relbias) {
        int qrow = q0 + wave * 16 + j + g * 8;
        int i0 = qrow - (kv0 + r) + 1023;
        int i1 = qrow - (kv0 + 16 + r) + 1023;
        i0 = i0 < 0 ? 0 : (i0 > 2046 ? 2046 : i0);
        i1 = i1 < 0 ? 0 : (i1 > 2046 ? 2046 : i1);
        s0 += relbias[i0];
        s1 += relbias[i1];
      }
      float mx = fmaxf(s0, s1);
#pragma unroll
      for (int off = 1; off < 16; off <<= 1)
        mx = fmaxf(mx, __shfl_xor(mx, off, 32));
      float mnew = fmaxf(mrun[j], mx);
      float corr = __expf(mrun[j] - mnew);
      float p0 = __expf(s0 - mnew);
      float p1 = __expf(s1 - mnew);
      float sum = p0 + p1;
#pragma unroll
      for (int off = 1; off < 16; off <<= 1)
        sum += __shfl_xor(sum, off, 32);
      lrun[j] = lrun[j] * corr + sum;
      mrun[j] = mnew;
#pragma unroll
      for (int nt = 0; nt < 4; ++nt) o[nt][j] *= corr;
      sP[wave][j + g * 8][r]      = f2bf(p0);
      sP[wave][j + g * 8][r + 16] = f2bf(p1);
    }
    __syncthreads();

    // P(16x32) as A-frag, V tiles as B-frags -> accumulate O
    FragBF pf;
    {
      const unsigned* row = (const unsigned*)&sP[wave][r][0];
      pf.q[0] = *(const uint4v*)(row + g * 4);
      pf.q[1] = *(const uint4v*)(row + 8 + g * 4);
    }
#pragma unroll
    for (int nt = 0; nt < 4; ++nt) {
      const unsigned* row = (const unsigned*)&sV[nt * 16 + r][0];
      FragBF vf;
      vf.q[0] = *(const uint4v*)(row + g * 4);
      vf.q[1] = *(const uint4v*)(row + 8 + g * 4);
      o[nt] = wmma_bf16(pf.v, vf.v, o[nt]);
    }
  }

#pragma unroll
  for (int j = 0; j < 8; ++j) {
    const float inv = 1.0f / lrun[j];
    const int qq = q0 + wave * 16 + j + g * 8;
#pragma unroll
    for (int nt = 0; nt < 4; ++nt)
      out[((size_t)bI * S + qq) * ostride + hI * 64 + nt * 16 + r] = o[nt][j] * inv;
  }
}

// ---------------- residual add + LayerNorm (D=1024, one block per row) ----------------
__global__ __launch_bounds__(256)
void sab_addln(const float* __restrict__ a, const float* __restrict__ bsrc,
               const float* __restrict__ gamma, const float* __restrict__ beta,
               float* __restrict__ out) {
  const int D = 1024;
  const size_t row = blockIdx.x;
  const float* pa = a + row * D;
  const float* pb = bsrc + row * D;
  float* po = out + row * D;
  const int tid = threadIdx.x;

  float v[4];
  float s = 0.f;
#pragma unroll
  for (int i = 0; i < 4; ++i) { v[i] = pa[tid + 256 * i] + pb[tid + 256 * i]; s += v[i]; }

  __shared__ float red[8];
#pragma unroll
  for (int off = 16; off > 0; off >>= 1) s += __shfl_xor(s, off, 32);
  if ((tid & 31) == 0) red[tid >> 5] = s;
  __syncthreads();
  float tot = 0.f;
#pragma unroll
  for (int w = 0; w < 8; ++w) tot += red[w];
  const float mean = tot * (1.0f / 1024.0f);

  float s2 = 0.f;
#pragma unroll
  for (int i = 0; i < 4; ++i) { float d = v[i] - mean; s2 += d * d; }
  __syncthreads();
#pragma unroll
  for (int off = 16; off > 0; off >>= 1) s2 += __shfl_xor(s2, off, 32);
  if ((tid & 31) == 0) red[tid >> 5] = s2;
  __syncthreads();
  float tv = 0.f;
#pragma unroll
  for (int w = 0; w < 8; ++w) tv += red[w];
  const float rstd = rsqrtf(tv * (1.0f / 1024.0f) + 1e-5f);

#pragma unroll
  for (int i = 0; i < 4; ++i) {
    int idx = tid + 256 * i;
    po[idx] = (v[i] - mean) * rstd * gamma[idx] + beta[idx];
  }
}

// ---------------- launcher ----------------
extern "C" void kernel_launch(void* const* d_in, const int* in_sizes, int n_in,
                              void* d_out, int out_size, void* d_ws, size_t ws_size,
                              hipStream_t stream) {
  (void)in_sizes; (void)n_in; (void)out_size; (void)ws_size;
  constexpr int B = 4, S = 1024, D = 1024, H = 16, DFF = 4096;

  const float* hidden = (const float*)d_in[0];
  const float* source = (const float*)d_in[1];
  const int*   tstep  = (const int*)d_in[2];
  const float* stepe  = (const float*)d_in[3];
  const float* relb   = (const float*)d_in[4];
  const float* Wqkv = (const float*)d_in[5];  const float* bqkv = (const float*)d_in[6];
  const float* Wo   = (const float*)d_in[7];  const float* bo   = (const float*)d_in[8];
  const float* Wcq  = (const float*)d_in[9];  const float* bcq  = (const float*)d_in[10];
  const float* Wckv = (const float*)d_in[11]; const float* bckv = (const float*)d_in[12];
  const float* Wco  = (const float*)d_in[13]; const float* bco  = (const float*)d_in[14];
  const float* W1m  = (const float*)d_in[15]; const float* b1   = (const float*)d_in[16];
  const float* W2m  = (const float*)d_in[17]; const float* b2   = (const float*)d_in[18];
  const float* g_self  = (const float*)d_in[19]; const float* bt_self  = (const float*)d_in[20];
  const float* g_cross = (const float*)d_in[21]; const float* bt_cross = (const float*)d_in[22];
  const float* g_ffn   = (const float*)d_in[23]; const float* bt_ffn   = (const float*)d_in[24];
  float* out = (float*)d_out;

  const size_t NT = (size_t)B * S;       // 4096 tokens
  float* W0b = (float*)d_ws;             // (B,S,D)   current hidden
  float* W1b = W0b + NT * D;             // (B,S,DFF) also qkv(3D)/ckv(2D)
  float* W2b = W1b + NT * DFF;           // (B,S,D)
  float* W3b = W2b + NT * D;             // (B,S,D)
  const int M = (int)NT;

  // 1. h = hidden + step_embed[clamp(step)]
  sab_add_step<<<dim3((B * S * D) / 256), dim3(256), 0, stream>>>(
      hidden, stepe, tstep, W0b, B * S * D, D);

  // 2. qkv = h @ Wqkv + bqkv   (M x 3D)
  sab_gemm_wmma<<<dim3(3 * D / 128, M / 128), dim3(256), 0, stream>>>(
      W0b, Wqkv, bqkv, W1b, M, 3 * D, D, 0);

  // 3. self-attention with rel-pos bias -> W2b (B,S,D)
  sab_attn_wmma<<<dim3(S / 64, H, B), dim3(128), 0, stream>>>(
      W1b, 3 * D, W1b + D, 3 * D, W1b + 2 * D, 3 * D, relb, W2b, D, S);

  // 4. so = attn @ Wo + bo -> W3b
  sab_gemm_wmma<<<dim3(D / 128, M / 128), dim3(256), 0, stream>>>(
      W2b, Wo, bo, W3b, M, D, D, 0);

  // 5. h = LN(h + so)
  sab_addln<<<dim3(M), dim3(256), 0, stream>>>(W0b, W3b, g_self, bt_self, W0b);

  // 6. cq = h @ Wcq + bcq -> W2b
  sab_gemm_wmma<<<dim3(D / 128, M / 128), dim3(256), 0, stream>>>(
      W0b, Wcq, bcq, W2b, M, D, D, 0);

  // 7. ckv = source @ Wckv + bckv -> W1b (B,S,2D)
  sab_gemm_wmma<<<dim3(2 * D / 128, M / 128), dim3(256), 0, stream>>>(
      source, Wckv, bckv, W1b, M, 2 * D, D, 0);

  // 8. cross-attention (no bias) -> W3b
  sab_attn_wmma<<<dim3(S / 64, H, B), dim3(128), 0, stream>>>(
      W2b, D, W1b, 2 * D, W1b + D, 2 * D, nullptr, W3b, D, S);

  // 9. co = cross @ Wco + bco -> W2b
  sab_gemm_wmma<<<dim3(D / 128, M / 128), dim3(256), 0, stream>>>(
      W3b, Wco, bco, W2b, M, D, D, 0);

  // 10. h = LN(h + co)
  sab_addln<<<dim3(M), dim3(256), 0, stream>>>(W0b, W2b, g_cross, bt_cross, W0b);

  // 11. ff = GELU(h @ W1 + b1) -> W1b (B,S,DFF)
  sab_gemm_wmma<<<dim3(DFF / 128, M / 128), dim3(256), 0, stream>>>(
      W0b, W1m, b1, W1b, M, DFF, D, 1);

  // 12. ff2 = ff @ W2 + b2 -> W2b
  sab_gemm_wmma<<<dim3(D / 128, M / 128), dim3(256), 0, stream>>>(
      W1b, W2m, b2, W2b, M, D, DFF, 0);

  // 13. out = LN(h + ff2)
  sab_addln<<<dim3(M), dim3(256), 0, stream>>>(W0b, W2b, g_ffn, bt_ffn, out);
}
